// GNNCWT2D_Mk11_1sec_63651415327484
// MI455X (gfx1250) — matrix operand order repaired and checked by reading, not compiled, not run on hardware
//
// GNN-on-CWT pipeline for MI455X (gfx1250, wave32, WMMA bf16 16x16x32).
//
//   pool (f32->bf16, mean-of-5)                      1.25 GB read (BW floor)
//   GEMM 800->512 +bias+relu (WMMA bf16, async-LDS double-buffered)
//   per-electrode BN folded ALGEBRAICALLY into the next GEMM's epilogue:
//     (s_e*A + t_e) @ W = s_e*(A@W) + t_e*colsum(W)
//   GEMM 512->256, 256->128 (row-BN in epilogue)
//   graph-conv prep: BN-apply + 60-edge aggregate -> Acat=[agg|x]
//   GEMM 256->64 (= agg@Wrel + x@Wroot + brel), per-column stats
//   graph-conv 2, GEMM 128->64, stats
//   head: col-BN + segment_max(19) + 64->32 relu -> 32->2
//
// GEMM staging uses global_load_async_to_lds_b128 with ping-pong LDS buffers
// and s_wait_asynccnt-based pipelining (ASYNCcnt retires in order).

#include <hip/hip_runtime.h>
#include <hip/hip_bf16.h>
#include <cstdint>

typedef unsigned short u16;
typedef unsigned int   u32;

typedef __bf16 bf16x16 __attribute__((ext_vector_type(16)));
typedef float  f32x8   __attribute__((ext_vector_type(8)));

#define E_     19
#define B_     4096
#define M_     (B_ * E_)   /* 77824 = 608 * 128 */
#define NEDGE_ 60

__device__ __forceinline__ u16 f2bf(float f) {
  u32 u = __builtin_bit_cast(u32, f);
  u32 r = u + 0x7FFFu + ((u >> 16) & 1u);   // round-to-nearest-even
  return (u16)(r >> 16);
}
__device__ __forceinline__ float bf2f(u16 h) {
  u32 u = ((u32)h) << 16;
  return __builtin_bit_cast(float, u);
}

union Frag { uint4 q[2]; bf16x16 v; };

// ---------------------------------------------------------------- pooling
__global__ __launch_bounds__(256) void pool_kernel(const float* __restrict__ x,
                                                   u16* __restrict__ h0) {
  int row = blockIdx.x;
  const float* xr = x + (size_t)row * 4000;
  u16* hr = h0 + (size_t)row * 800;
  for (int o = threadIdx.x; o < 800; o += 256) {
    int f = o / 20, gi = o % 20;
    const float* p = xr + f * 100 + gi * 5;
    float s = p[0] + p[1] + p[2] + p[3] + p[4];
    hr[o] = f2bf(s * 0.2f);
  }
}

// ---------------------------------------------------------------- WMMA GEMM
// A: (M_, K) bf16 row-major.  Wt: (N, K) bf16 (pre-transposed weights).
// out = relu( BNrow(A) @ W + bias ), BNrow folded into epilogue via wsum.
// Block: 256 thr = 8 waves.  Tile 128 x BN (BN = 128 if N>=128 else 64),
// K-step 32.  Async-LDS staging, ping-pong double buffer.
template <int K, int N, bool ROWBN>
__global__ __launch_bounds__(256) void gemm_kernel(
    const u16* __restrict__ A, const u16* __restrict__ Wt,
    const float* __restrict__ bias, const float* __restrict__ wsum,
    const float2* __restrict__ ss, u16* __restrict__ out) {
  constexpr int BN  = (N >= 128) ? 128 : 64;
  constexpr int WN  = BN / 2;        // N-span per wave (2 waves in N)
  constexpr int JT  = WN / 16;       // WMMA tiles per wave in N (4 or 2)
  constexpr int BCH = BN / 64;       // B-tile 16B chunks per thread (2 or 1)
  constexpr int NPER = 2 + BCH;      // async instrs per wave per tile batch
  constexpr int T   = K / 32;

  __shared__ __align__(16) u16 sA[2][128 * 40];   // padded stride 40 elems
  __shared__ __align__(16) u16 sB[2][BN * 40];
  __shared__ float2 ssl[19];

  const int tid  = threadIdx.x;
  const int lane = tid & 31, wv = tid >> 5;
  const int hf   = lane >> 4, l16 = lane & 15;
  const int wm   = wv & 3,    wn  = wv >> 2;
  const int bm   = blockIdx.x * 128;
  const int bn   = blockIdx.y * BN;

  if (ROWBN) { if (tid < 19) ssl[tid] = ss[tid]; }  // visible after 1st barrier

  // issue one tile batch (A 128x32, B BNx32) via async DMA into LDS buffer bs
  auto issue = [&](int kt, int bs) {
#pragma unroll
    for (int i = 0; i < 2; ++i) {                       // A: 2 chunks/thread
      int c = tid + i * 256, r = c >> 2, ci = c & 3;
      u32 lo = (u32)(uintptr_t)&sA[bs][r * 40 + ci * 8];
      const u16* gp = A + (size_t)(bm + r) * K + kt + ci * 8;
      asm volatile("global_load_async_to_lds_b128 %0, %1, off"
                   :: "v"(lo), "v"(gp) : "memory");
    }
#pragma unroll
    for (int i = 0; i < BCH; ++i) {                     // B: BCH chunks/thread
      int c = tid + i * 256, r = c >> 2, ci = c & 3;
      u32 lo = (u32)(uintptr_t)&sB[bs][r * 40 + ci * 8];
      const u16* gp = Wt + (size_t)(bn + r) * K + kt + ci * 8;
      asm volatile("global_load_async_to_lds_b128 %0, %1, off"
                   :: "v"(lo), "v"(gp) : "memory");
    }
  };

  f32x8 acc[2][JT] = {};

  issue(0, 0);
  for (int t = 0; t < T; ++t) {
    if (t + 1 < T) {
      issue((t + 1) * 32, (t + 1) & 1);
      // batch t fully retired once <= NPER (newest batch) remain outstanding
      asm volatile("s_wait_asynccnt %0" :: "i"(NPER) : "memory");
    } else {
      asm volatile("s_wait_asynccnt 0" ::: "memory");
    }
    __syncthreads();

    const u16* cA = sA[t & 1];
    const u16* cB = sB[t & 1];
    Frag a[2], b[JT];
#pragma unroll
    for (int i = 0; i < 2; ++i) {
      int rA = wm * 32 + i * 16 + l16;
      a[i].q[0] = *(const uint4*)&cA[rA * 40 + hf * 8];       // K hf*8..+7
      a[i].q[1] = *(const uint4*)&cA[rA * 40 + hf * 8 + 16];  // K 16+hf*8..
    }
#pragma unroll
    for (int j = 0; j < JT; ++j) {
      int cBr = wn * WN + j * 16 + l16;
      b[j].q[0] = *(const uint4*)&cB[cBr * 40 + hf * 8];      // K hf*8..+7
      b[j].q[1] = *(const uint4*)&cB[cBr * 40 + hf * 8 + 16]; // K 16+hf*8..
    }
#pragma unroll
    for (int i = 0; i < 2; ++i)
#pragma unroll
      for (int j = 0; j < JT; ++j)
        acc[i][j] = __builtin_amdgcn_wmma_f32_16x16x32_bf16(
            false, a[i].v, false, b[j].v, (short)0, acc[i][j], false, false);
    __syncthreads();
  }

  // ---- epilogue: row-BN (scale/shift via wsum), +bias, relu, store bf16
#pragma unroll
  for (int i = 0; i < 2; ++i)
#pragma unroll
    for (int j = 0; j < JT; ++j) {
      int col = bn + wn * WN + j * 16 + l16;
      float bsc = bias[col];
      float wsc = ROWBN ? wsum[col] : 0.f;
#pragma unroll
      for (int v = 0; v < 8; ++v) {
        int row = bm + wm * 32 + i * 16 + hf * 8 + v;
        float y = acc[i][j][v];
        if (ROWBN) {
          float2 p = ssl[row % 19];
          y = p.x * y + p.y * wsc;
        }
        y += bsc;
        out[(size_t)row * N + col] = f2bf(fmaxf(y, 0.0f));
      }
    }
}

// ------------------------------------------------- BN stats, electrode mode
template <int N>
__global__ __launch_bounds__(256) void statsE_kernel(const u16* __restrict__ h,
                                                     float* __restrict__ gs,
                                                     float* __restrict__ gq) {
  __shared__ float es[19], eq[19];
  int tid = threadIdx.x;
  if (tid < 19) { es[tid] = 0.f; eq[tid] = 0.f; }
  __syncthreads();
  int lane = tid & 31, wv = tid >> 5;
  int row = blockIdx.x * 8 + wv;
  const u16* p = h + (size_t)row * N;
  float s = 0.f, q = 0.f;
  for (int i = lane; i < N; i += 32) { float y = bf2f(p[i]); s += y; q += y * y; }
  int e = row % 19;
  atomicAdd(&es[e], s);
  atomicAdd(&eq[e], q);
  __syncthreads();
  if (tid < 19) {
    int slot = blockIdx.x & 63;
    atomicAdd(&gs[tid * 64 + slot], es[tid]);
    atomicAdd(&gq[tid * 64 + slot], eq[tid]);
  }
}

// ------------------------------------------------- BN stats, column mode (64)
__global__ __launch_bounds__(256) void statsC_kernel(const u16* __restrict__ h,
                                                     float* __restrict__ gs,
                                                     float* __restrict__ gq) {
  __shared__ float cs[64], cq[64];
  int tid = threadIdx.x;
  if (tid < 64) { cs[tid] = 0.f; cq[tid] = 0.f; }
  __syncthreads();
  int col = tid & 63, ro = tid >> 6;
  size_t base = (size_t)blockIdx.x * 64;
  float s = 0.f, q = 0.f;
  for (int r = ro; r < 64; r += 4) {
    float y = bf2f(h[(base + r) * 64 + col]);
    s += y; q += y * y;
  }
  atomicAdd(&cs[col], s);
  atomicAdd(&cq[col], q);
  __syncthreads();
  if (tid < 64) {
    int slot = blockIdx.x & 63;
    atomicAdd(&gs[tid * 64 + slot], cs[tid]);
    atomicAdd(&gq[tid * 64 + slot], cq[tid]);
  }
}

// ---------------------------------------------- finalize BN -> (scale, shift)
__global__ void finalize_kernel(const float* __restrict__ gs,
                                const float* __restrict__ gq,
                                const float* __restrict__ g,
                                const float* __restrict__ be,
                                float2* __restrict__ ss, int n, float cnt) {
  int t = threadIdx.x;
  if (t < n) {
    float s = 0.f, q = 0.f;
    for (int i = 0; i < 64; ++i) { s += gs[t * 64 + i]; q += gq[t * 64 + i]; }
    float m  = s / cnt;
    float v  = q / cnt - m * m;                 // biased var (ddof=0)
    float sc = g[t] * rsqrtf(v + 1e-5f);
    ss[t] = make_float2(sc, be[t] - m * sc);
  }
}

// ---------------------------------------------- graph-conv prep (per graph)
template <int F, bool ROWMODE>
__global__ __launch_bounds__(F) void prep_kernel(
    const u16* __restrict__ hin, const float2* __restrict__ ss,
    const int* __restrict__ src, const int* __restrict__ dst,
    const float* __restrict__ ew, u16* __restrict__ acat) {
  __shared__ float xn[19 * F];
  __shared__ float ag[19 * F];
  int g = blockIdx.x, f = threadIdx.x;
  int base = g * 19;
#pragma unroll
  for (int v = 0; v < 19; ++v) {
    float y  = bf2f(hin[(size_t)(base + v) * F + f]);
    float2 p = ROWMODE ? ss[v] : ss[f];
    xn[v * F + f] = y * p.x + p.y;
    ag[v * F + f] = 0.f;
  }
  for (int e = 0; e < NEDGE_; ++e) {
    int sl = src[g * NEDGE_ + e] - base;
    int dl = dst[g * NEDGE_ + e] - base;
    float w = ew[e];
    ag[dl * F + f] += xn[sl * F + f] * w;   // thread owns column f: race-free
  }
#pragma unroll
  for (int v = 0; v < 19; ++v) {
    size_t row = base + v;
    acat[row * (2 * F) + f]     = f2bf(ag[v * F + f]);
    acat[row * (2 * F) + F + f] = f2bf(xn[v * F + f]);
  }
}

// ---------------------------------------------- head: BN + seg-max + MLP
__global__ __launch_bounds__(256) void head_kernel(
    const u16* __restrict__ h, const float2* __restrict__ ss,
    const float* __restrict__ W5, const float* __restrict__ b5,
    const float* __restrict__ W6, const float* __restrict__ b6,
    float* __restrict__ out) {
  __shared__ float pooled[8][64];
  __shared__ float hidden[8][32];
  int tid = threadIdx.x, lane = tid & 31, wv = tid >> 5;
  int g = blockIdx.x * 8 + wv;
  size_t base = (size_t)g * 19;
#pragma unroll
  for (int hv = 0; hv < 2; ++hv) {
    int c = lane + hv * 32;
    float2 p = ss[c];
    float m = -3.402823466e38f;
    for (int v = 0; v < 19; ++v)
      m = fmaxf(m, bf2f(h[(base + v) * 64 + c]) * p.x + p.y);
    pooled[wv][c] = m;
  }
  __syncthreads();
  {
    float acc = b5[lane];
    for (int c = 0; c < 64; ++c) acc += pooled[wv][c] * W5[c * 32 + lane];
    hidden[wv][lane] = fmaxf(acc, 0.f);
  }
  __syncthreads();
  if (lane < 2) {
    float o = b6[lane];
    for (int j = 0; j < 32; ++j) o += hidden[wv][j] * W6[j * 2 + lane];
    out[(size_t)g * 2 + lane] = o;
  }
}

// ---------------------------------------------- weight prep kernels
__global__ void convt_kernel(const float* __restrict__ W, u16* __restrict__ Wt,
                             int K, int N) {
  int idx = blockIdx.x * 256 + threadIdx.x;
  if (idx < K * N) {
    int k = idx / N, n = idx % N;
    Wt[(size_t)n * K + k] = f2bf(W[idx]);    // (K,N) f32 -> (N,K) bf16
  }
}
__global__ void wsum_kernel(const float* __restrict__ W, float* __restrict__ ws,
                            int K, int N) {
  int n = blockIdx.x * 256 + threadIdx.x;
  if (n < N) {
    float s = 0.f;
    for (int k = 0; k < K; ++k) s += W[(size_t)k * N + n];
    ws[n] = s;                               // colsum(W) for epilogue BN fold
  }
}
__global__ void packg_kernel(const float* __restrict__ Wrel,
                             const float* __restrict__ Wroot,
                             u16* __restrict__ Wt, int F) {
  int idx = blockIdx.x * 256 + threadIdx.x;
  int total = 64 * 2 * F;
  if (idx < total) {
    int n = idx / (2 * F), k = idx % (2 * F);
    float v = (k < F) ? Wrel[(size_t)k * 64 + n] : Wroot[(size_t)(k - F) * 64 + n];
    Wt[idx] = f2bf(v);                       // Wt (64, 2F) = [Wrel;Wroot]^T
  }
}

// ================================================================ launcher
extern "C" void kernel_launch(void* const* d_in, const int* in_sizes, int n_in,
                              void* d_out, int out_size, void* d_ws,
                              size_t ws_size, hipStream_t stream) {
  const float* x     = (const float*)d_in[0];
  const int*   eidx  = (const int*)d_in[1];
  const float* W2 = (const float*)d_in[3];  const float* b2 = (const float*)d_in[4];
  const float* W3 = (const float*)d_in[5];  const float* b3 = (const float*)d_in[6];
  const float* W4 = (const float*)d_in[7];  const float* b4 = (const float*)d_in[8];
  const float* ew1   = (const float*)d_in[9];
  const float* Wrel1 = (const float*)d_in[10]; const float* brel1 = (const float*)d_in[11];
  const float* Wroot1= (const float*)d_in[12];
  const float* ew2   = (const float*)d_in[13];
  const float* Wrel2 = (const float*)d_in[14]; const float* brel2 = (const float*)d_in[15];
  const float* Wroot2= (const float*)d_in[16];
  const float* W5 = (const float*)d_in[17]; const float* b5 = (const float*)d_in[18];
  const float* W6 = (const float*)d_in[19]; const float* b6 = (const float*)d_in[20];
  const float* g3 = (const float*)d_in[21]; const float* be3 = (const float*)d_in[22];
  const float* g4 = (const float*)d_in[23]; const float* be4 = (const float*)d_in[24];
  const float* g5 = (const float*)d_in[25]; const float* be5 = (const float*)d_in[26];
  const float* g6 = (const float*)d_in[27]; const float* be6 = (const float*)d_in[28];
  const float* g7 = (const float*)d_in[29]; const float* be7 = (const float*)d_in[30];

  const int* esrc = eidx;
  const int* edst = eidx + (size_t)B_ * NEDGE_;

  char* ws = (char*)d_ws;
  size_t off = 0;
  auto alloc = [&](size_t bytes) {
    size_t o = off; off = (off + bytes + 255) & ~(size_t)255; return o;
  };

  // Region A (124.5 MB) and B (79.7 MB), lifetime-based reuse.
  size_t offA = alloc((size_t)M_ * 800 * 2);   // h0 / h2 / Acat1 / g1 / Acat2 / g2
  size_t offB = alloc((size_t)M_ * 512 * 2);   // h1 / h3
  u16* h0    = (u16*)(ws + offA);
  u16* h1    = (u16*)(ws + offB);
  u16* h2    = (u16*)(ws + offA);                     // h0 dead
  u16* h3    = (u16*)(ws + offB);                     // h1 dead
  u16* acat1 = (u16*)(ws + offA);                     // h2 dead
  u16* g1b   = (u16*)(ws + offA + 41943040);
  u16* acat2 = (u16*)(ws + offA + 54525952);
  u16* g2b   = (u16*)(ws + offA + 75497472);

  u16* wt2  = (u16*)(ws + alloc((size_t)512 * 800 * 2));
  u16* wt3  = (u16*)(ws + alloc((size_t)256 * 512 * 2));
  u16* wt4  = (u16*)(ws + alloc((size_t)128 * 256 * 2));
  u16* wtg1 = (u16*)(ws + alloc((size_t)64 * 256 * 2));
  u16* wtg2 = (u16*)(ws + alloc((size_t)64 * 128 * 2));
  float* wsum3 = (float*)(ws + alloc(256 * 4));
  float* wsum4 = (float*)(ws + alloc(128 * 4));

  size_t statsBase = off;
  float* sE1s = (float*)(ws + alloc(19 * 64 * 4));
  float* sE1q = (float*)(ws + alloc(19 * 64 * 4));
  float* sE2s = (float*)(ws + alloc(19 * 64 * 4));
  float* sE2q = (float*)(ws + alloc(19 * 64 * 4));
  float* sE3s = (float*)(ws + alloc(19 * 64 * 4));
  float* sE3q = (float*)(ws + alloc(19 * 64 * 4));
  float* sC1s = (float*)(ws + alloc(64 * 64 * 4));
  float* sC1q = (float*)(ws + alloc(64 * 64 * 4));
  float* sC2s = (float*)(ws + alloc(64 * 64 * 4));
  float* sC2q = (float*)(ws + alloc(64 * 64 * 4));
  size_t statsBytes = off - statsBase;

  float2* ssE1 = (float2*)(ws + alloc(19 * 8));
  float2* ssE2 = (float2*)(ws + alloc(19 * 8));
  float2* ssE3 = (float2*)(ws + alloc(19 * 8));
  float2* ssC1 = (float2*)(ws + alloc(64 * 8));
  float2* ssC2 = (float2*)(ws + alloc(64 * 8));

  // weight conversion + stats zeroing (capture-safe)
  convt_kernel<<<(800 * 512 + 255) / 256, 256, 0, stream>>>(W2, wt2, 800, 512);
  convt_kernel<<<(512 * 256 + 255) / 256, 256, 0, stream>>>(W3, wt3, 512, 256);
  convt_kernel<<<(256 * 128 + 255) / 256, 256, 0, stream>>>(W4, wt4, 256, 128);
  wsum_kernel<<<1, 256, 0, stream>>>(W3, wsum3, 512, 256);
  wsum_kernel<<<1, 256, 0, stream>>>(W4, wsum4, 256, 128);
  packg_kernel<<<(64 * 256 + 255) / 256, 256, 0, stream>>>(Wrel1, Wroot1, wtg1, 128);
  packg_kernel<<<(64 * 128 + 255) / 256, 256, 0, stream>>>(Wrel2, Wroot2, wtg2, 64);
  hipMemsetAsync(ws + statsBase, 0, statsBytes, stream);

  pool_kernel<<<M_, 256, 0, stream>>>(x, h0);

  gemm_kernel<800, 512, false><<<dim3(M_ / 128, 4), 256, 0, stream>>>(
      h0, wt2, b2, nullptr, nullptr, h1);
  statsE_kernel<512><<<M_ / 8, 256, 0, stream>>>(h1, sE1s, sE1q);
  finalize_kernel<<<1, 64, 0, stream>>>(sE1s, sE1q, g3, be3, ssE1, 19,
                                        (float)((size_t)B_ * 512));

  gemm_kernel<512, 256, true><<<dim3(M_ / 128, 2), 256, 0, stream>>>(
      h1, wt3, b3, wsum3, ssE1, h2);
  statsE_kernel<256><<<M_ / 8, 256, 0, stream>>>(h2, sE2s, sE2q);
  finalize_kernel<<<1, 64, 0, stream>>>(sE2s, sE2q, g4, be4, ssE2, 19,
                                        (float)((size_t)B_ * 256));

  gemm_kernel<256, 128, true><<<dim3(M_ / 128, 1), 256, 0, stream>>>(
      h2, wt4, b4, wsum4, ssE2, h3);
  statsE_kernel<128><<<M_ / 8, 256, 0, stream>>>(h3, sE3s, sE3q);
  finalize_kernel<<<1, 64, 0, stream>>>(sE3s, sE3q, g5, be5, ssE3, 19,
                                        (float)((size_t)B_ * 128));

  prep_kernel<128, true><<<B_, 128, 0, stream>>>(h3, ssE3, esrc, edst, ew1, acat1);
  gemm_kernel<256, 64, false><<<dim3(M_ / 128, 1), 256, 0, stream>>>(
      acat1, wtg1, brel1, nullptr, nullptr, g1b);
  statsC_kernel<<<M_ / 64, 256, 0, stream>>>(g1b, sC1s, sC1q);
  finalize_kernel<<<1, 64, 0, stream>>>(sC1s, sC1q, g6, be6, ssC1, 64, (float)M_);

  prep_kernel<64, false><<<B_, 64, 0, stream>>>(g1b, ssC1, esrc, edst, ew2, acat2);
  gemm_kernel<128, 64, false><<<dim3(M_ / 128, 1), 256, 0, stream>>>(
      acat2, wtg2, brel2, nullptr, nullptr, g2b);
  statsC_kernel<<<M_ / 64, 256, 0, stream>>>(g2b, sC2s, sC2q);
  finalize_kernel<<<1, 64, 0, stream>>>(sC2s, sC2q, g7, be7, ssC2, 64, (float)M_);

  head_kernel<<<B_ / 8, 256, 0, stream>>>(g2b, ssC2, W5, b5, W6, b6, (float*)d_out);
}
